// MultiheadCubeAttention_84473416778046
// MI455X (gfx1250) — matrix operand order repaired and checked by reading, not compiled, not run on hardware
//
#include <hip/hip_runtime.h>
#include <math.h>

// ---------------------------------------------------------------------------
// MultiheadCubeAttention for MI455X (gfx1250, wave32, WMMA bf16 w/ f32 accum)
//
// S=512, D=256, H=8, DH=32.
// Pipeline:
//   K1: qp = q @ Wq^T (f32) + bf16 copy     (tiny)
//   K1b: k,v -> bf16                        (tiny, kills in-loop cvts)
//   K2: Pk/Pv = qp @ W_hyper^T  -> bf16     (WMMA bf16; 4 acc chains/wave)
//   K3: fused per-query cube attention      (WMMA bf16 + online softmax;
//       Pk[i],Pv[i] staged to LDS via async-to-LDS when available)
// ---------------------------------------------------------------------------

#define SEQ 512
#define DM 256
#define NH 8
#define DH 32
#define LDSROW 264  // 256 bf16 + 8 pad -> 528B row stride (16B aligned, spreads banks)

#if __has_builtin(__builtin_amdgcn_global_load_async_to_lds_b128) && \
    __has_builtin(__builtin_amdgcn_s_wait_asynccnt)
#define USE_ASYNC_LDS 1
#else
#define USE_ASYNC_LDS 0
#endif

typedef __attribute__((ext_vector_type(16))) __bf16 v16bf;
typedef __attribute__((ext_vector_type(8)))  __bf16 bf16x8;
typedef __attribute__((ext_vector_type(8)))  float  v8f;
typedef __attribute__((ext_vector_type(4)))  int    v4i;

#if USE_ASYNC_LDS
typedef __attribute__((address_space(1))) v4i g_v4i;  // global int4
typedef __attribute__((address_space(3))) v4i l_v4i;  // LDS int4
#endif

union FragU { v16bf v; bf16x8 h[2]; __bf16 e[16]; };

__device__ __forceinline__ v8f wmma_bf16f32(v16bf a, v16bf b, v8f c) {
  return __builtin_amdgcn_wmma_f32_16x16x32_bf16(false, a, false, b, (short)0, c,
                                                 false, false);
}

// A-matrix 16x32 bf16 fragment. Lane half selects contiguous K half:
// lanes<16 K{0..15}, lanes>=16 K{16..31} (matches B fragment order).
__device__ __forceinline__ v16bf load_a_bf16(const __bf16* rp, int lane) {
  const int kb = (lane & 16) ? 16 : 0;
  FragU u;
  u.h[0] = *(const bf16x8*)(rp + kb);
  u.h[1] = *(const bf16x8*)(rp + kb + 8);
  return u.v;
}

// B-matrix 32x16 bf16 fragment: lane holds column N=lane%16 of B = row N of
// the row-major B^T source; 16 contiguous K at K-half 0/16 by lane half.
__device__ __forceinline__ v16bf load_b_bf16(const __bf16* rp, int lane) {
  const int kh = (lane & 16) ? 16 : 0;
  FragU u;
  u.h[0] = *(const bf16x8*)(rp + kh);
  u.h[1] = *(const bf16x8*)(rp + kh + 8);
  return u.v;
}

__device__ __forceinline__ v16bf load_b_f32(const float* rp, int lane) {
  const int kh = (lane & 16) ? 16 : 0;
  float4 c0 = *(const float4*)(rp + kh);
  float4 c1 = *(const float4*)(rp + kh + 4);
  float4 c2 = *(const float4*)(rp + kh + 8);
  float4 c3 = *(const float4*)(rp + kh + 12);
  FragU u;
  u.e[0] = (__bf16)c0.x; u.e[1] = (__bf16)c0.y; u.e[2]  = (__bf16)c0.z; u.e[3]  = (__bf16)c0.w;
  u.e[4] = (__bf16)c1.x; u.e[5] = (__bf16)c1.y; u.e[6]  = (__bf16)c1.z; u.e[7]  = (__bf16)c1.w;
  u.e[8] = (__bf16)c2.x; u.e[9] = (__bf16)c2.y; u.e[10] = (__bf16)c2.z; u.e[11] = (__bf16)c2.w;
  u.e[12]= (__bf16)c3.x; u.e[13]= (__bf16)c3.y; u.e[14] = (__bf16)c3.z; u.e[15] = (__bf16)c3.w;
  return u.v;
}

// ---------------------------------------------------------------------------
// K1: qp = q @ Wq^T  (also emit bf16 copy for the WMMA GEMMs)
// ---------------------------------------------------------------------------
__global__ void __launch_bounds__(256) qproj_kernel(
    const float* __restrict__ q, const float* __restrict__ Wq,
    float* __restrict__ qp, __bf16* __restrict__ qph) {
  const int i = blockIdx.x;
  const int j = threadIdx.x;
  const float4* qr = (const float4*)(q + i * DM);
  const float4* wr = (const float4*)(Wq + j * DM);
  float acc = 0.f;
#pragma unroll 8
  for (int t = 0; t < DM / 4; ++t) {
    float4 a = qr[t], b = wr[t];
    acc += a.x * b.x + a.y * b.y + a.z * b.z + a.w * b.w;
  }
  qp[i * DM + j] = acc;
  qph[i * DM + j] = (__bf16)acc;
}

// ---------------------------------------------------------------------------
// K1b: f32 -> bf16 bulk conversion (8 elems/thread)
// ---------------------------------------------------------------------------
__global__ void __launch_bounds__(256) cvt_bf16_kernel(
    const float* __restrict__ src, __bf16* __restrict__ dst, int n8) {
  const int t = blockIdx.x * 256 + threadIdx.x;
  if (t >= n8) return;
  float4 a = ((const float4*)src)[2 * t];
  float4 b = ((const float4*)src)[2 * t + 1];
  union { bf16x8 v; __bf16 e[8]; } u;
  u.e[0] = (__bf16)a.x; u.e[1] = (__bf16)a.y; u.e[2] = (__bf16)a.z; u.e[3] = (__bf16)a.w;
  u.e[4] = (__bf16)b.x; u.e[5] = (__bf16)b.y; u.e[6] = (__bf16)b.z; u.e[7] = (__bf16)b.w;
  ((bf16x8*)dst)[t] = u.v;
}

// ---------------------------------------------------------------------------
// K2: P[i, jk] = sum_m qp[i,m] * W[jk, m]   (M=512, N=65536, K=256) -> bf16
// One block per 16-wide n-tile; wave w owns i-tiles {4w..4w+3}: one B fragment
// (f32 + cvt, amortized 4x) feeds 4 independent WMMA chains.
// ---------------------------------------------------------------------------
__global__ void __launch_bounds__(256) hyper_gemm_kernel(
    const __bf16* __restrict__ qph, const float* __restrict__ W,
    __bf16* __restrict__ P) {
  const int lane = threadIdx.x & 31;
  const int wave = threadIdx.x >> 5;
  const int ntile = blockIdx.x;  // 0..4095
  const int itbase = wave * 4;   // 4 i-tiles per wave -> 32 total
  const size_t nrow = (size_t)(ntile * 16 + (lane & 15));
  const __bf16* arow = qph + (itbase * 16 + (lane & 15)) * DM;

  v8f acc0 = {}, acc1 = {}, acc2 = {}, acc3 = {};
#pragma unroll
  for (int ks = 0; ks < 8; ++ks) {
    v16bf b = load_b_f32(W + nrow * DM + ks * 32, lane);
    v16bf a0 = load_a_bf16(arow + ks * 32, lane);
    v16bf a1 = load_a_bf16(arow + 16 * DM + ks * 32, lane);
    v16bf a2 = load_a_bf16(arow + 32 * DM + ks * 32, lane);
    v16bf a3 = load_a_bf16(arow + 48 * DM + ks * 32, lane);
    acc0 = wmma_bf16f32(a0, b, acc0);
    acc1 = wmma_bf16f32(a1, b, acc1);
    acc2 = wmma_bf16f32(a2, b, acc2);
    acc3 = wmma_bf16f32(a3, b, acc3);
  }
  const int half = lane >> 4;
  const int col = lane & 15;
  v8f accs[4] = {acc0, acc1, acc2, acc3};
#pragma unroll
  for (int t = 0; t < 4; ++t) {
#pragma unroll
    for (int r = 0; r < 8; ++r) {
      const int m = (itbase + t) * 16 + r + 8 * half;
      P[(size_t)m * 65536 + ntile * 16 + col] = (__bf16)accs[t][r];
    }
  }
}

// ---------------------------------------------------------------------------
// K3: fused per-query cube attention. Block = query i, wave = head.
// ---------------------------------------------------------------------------
__global__ void __launch_bounds__(256) cube_attn_kernel(
    const float* __restrict__ qp, const __bf16* __restrict__ kmat,
    const __bf16* __restrict__ vmat, const __bf16* __restrict__ Pk,
    const __bf16* __restrict__ Pv, float* __restrict__ out) {
  const int i = blockIdx.x;
  const int tid = threadIdx.x;
  extern __shared__ __align__(16) char smem_raw[];
  __bf16* sPk = (__bf16*)smem_raw;
  __bf16* sPv = sPk + 256 * LDSROW;

  // Stage Pk[i], Pv[i] (bf16, 128KB each) into padded LDS rows.
  const __bf16* gPk = Pk + (size_t)i * 65536;
  const __bf16* gPv = Pv + (size_t)i * 65536;
#if USE_ASYNC_LDS
  for (int idx = tid; idx < 8192; idx += 256) {
    const int row = idx >> 5;
    const int c = (idx & 31) * 8;  // 16B chunk
    __builtin_amdgcn_global_load_async_to_lds_b128(
        (g_v4i*)(gPk + row * 256 + c),
        (l_v4i*)(sPk + row * LDSROW + c), 0, 0);
    __builtin_amdgcn_global_load_async_to_lds_b128(
        (g_v4i*)(gPv + row * 256 + c),
        (l_v4i*)(sPv + row * LDSROW + c), 0, 0);
  }
  __builtin_amdgcn_s_wait_asynccnt(0);
#else
  for (int idx = tid; idx < 8192; idx += 256) {
    const int row = idx >> 5;
    const int c = (idx & 31) * 8;
    *(uint4*)(sPk + row * LDSROW + c) = *(const uint4*)(gPk + row * 256 + c);
    *(uint4*)(sPv + row * LDSROW + c) = *(const uint4*)(gPv + row * 256 + c);
  }
#endif
  __syncthreads();

  const int lane = tid & 31;
  const int wave = tid >> 5;  // head index
  const int half = lane >> 4;
  const int col = lane & 15;
  const int jb = wave * DH;

  const float qw0 = qp[i * DM + jb + col];
  const float qw1 = qp[i * DM + jb + 16 + col];
  const float scale = 0.044194173824159216f;  // 1/sqrt(512): ref scales by kv_seq

  const __bf16* bk0 = sPk + (jb + col) * LDSROW;
  const __bf16* bk1 = sPk + (jb + 16 + col) * LDSROW;
  const __bf16* bv0 = sPv + (jb + col) * LDSROW;
  const __bf16* bv1 = sPv + (jb + 16 + col) * LDSROW;

  float rmax = -3.0e38f, rsum = 0.f, o0 = 0.f, o1 = 0.f;
  const int ntiles = (i >> 4) + 1;  // causal: keys 0..i

  for (int t = 0; t < ntiles; ++t) {
    const int l0 = t * 16;
    const __bf16* krow = kmat + (l0 + col) * DM;
    const __bf16* vrow = vmat + (l0 + col) * DM;
    if (t + 1 < ntiles) {
      __builtin_prefetch(kmat + (l0 + 16 + col) * DM, 0, 0);
      __builtin_prefetch(vmat + (l0 + 16 + col) * DM, 0, 0);
    }

    // ---- K cube tile: Kc[l, j] for this head's 2 j-chunks
    v16bf afrag[8];
#pragma unroll
    for (int ks = 0; ks < 8; ++ks) afrag[ks] = load_a_bf16(krow + ks * 32, lane);
    v8f ak0 = {}, ak1 = {};
#pragma unroll
    for (int ks = 0; ks < 8; ++ks) {
      ak0 = wmma_bf16f32(afrag[ks], load_b_bf16(bk0 + ks * 32, lane), ak0);
      ak1 = wmma_bf16f32(afrag[ks], load_b_bf16(bk1 + ks * 32, lane), ak1);
    }

    // ---- logits for this lane's 8 rows (M = r + 8*half)
    float lg[8];
#pragma unroll
    for (int r = 0; r < 8; ++r) {
      float p = qw0 * ak0[r] + qw1 * ak1[r];
      p += __shfl_xor(p, 1);
      p += __shfl_xor(p, 2);
      p += __shfl_xor(p, 4);
      p += __shfl_xor(p, 8);  // sum over the 16 j-columns in this half
      const int l = l0 + r + 8 * half;
      lg[r] = (l <= i) ? p * scale : -3.0e38f;
    }

    // ---- online softmax update (state replicated per lane, uniform per wave)
    float tmax = lg[0];
#pragma unroll
    for (int r = 1; r < 8; ++r) tmax = fmaxf(tmax, lg[r]);
    tmax = fmaxf(tmax, __shfl_xor(tmax, 16));
    const float newmax = fmaxf(rmax, tmax);
    const float alpha = __expf(rmax - newmax);
    float pr[8];
    float tsum = 0.f;
#pragma unroll
    for (int r = 0; r < 8; ++r) {
      pr[r] = __expf(lg[r] - newmax);
      tsum += pr[r];
    }
    tsum += __shfl_xor(tsum, 16);
    rsum = rsum * alpha + tsum;
    rmax = newmax;
    o0 *= alpha;
    o1 *= alpha;

    // ---- V cube tile + PV accumulate (probs already match D-row layout)
#pragma unroll
    for (int ks = 0; ks < 8; ++ks) afrag[ks] = load_a_bf16(vrow + ks * 32, lane);
    v8f av0 = {}, av1 = {};
#pragma unroll
    for (int ks = 0; ks < 8; ++ks) {
      av0 = wmma_bf16f32(afrag[ks], load_b_bf16(bv0 + ks * 32, lane), av0);
      av1 = wmma_bf16f32(afrag[ks], load_b_bf16(bv1 + ks * 32, lane), av1);
    }
#pragma unroll
    for (int r = 0; r < 8; ++r) {
      o0 += pr[r] * av0[r];
      o1 += pr[r] * av1[r];
    }
  }

  // combine the two row-halves, normalize, store
  o0 += __shfl_xor(o0, 16);
  o1 += __shfl_xor(o1, 16);
  const float inv = 1.0f / rsum;
  if (half == 0)
    out[i * DM + jb + col] = o0 * inv;
  else
    out[i * DM + jb + 16 + col] = o1 * inv;
}

// ---------------------------------------------------------------------------
extern "C" void kernel_launch(void* const* d_in, const int* in_sizes, int n_in,
                              void* d_out, int out_size, void* d_ws,
                              size_t ws_size, hipStream_t stream) {
  const float* q = (const float*)d_in[0];
  const float* k = (const float*)d_in[1];
  const float* v = (const float*)d_in[2];
  const float* Wq = (const float*)d_in[3];
  const float* Wk = (const float*)d_in[4];
  const float* Wv = (const float*)d_in[5];
  // d_in[6] = tril mask: applied analytically (l <= i) in cube_attn_kernel.
  float* out = (float*)d_out;

  char* ws = (char*)d_ws;
  float* qp = (float*)ws;                               // 512 KB
  __bf16* qph = (__bf16*)(ws + 524288);                 // 256 KB
  __bf16* kbf = (__bf16*)(ws + 786432);                 // 256 KB
  __bf16* vbf = (__bf16*)(ws + 1048576);                // 256 KB
  __bf16* Pk = (__bf16*)(ws + 1310720);                 // 64 MB
  __bf16* Pv = (__bf16*)(ws + 1310720 + 67108864);      // 64 MB

  qproj_kernel<<<512, 256, 0, stream>>>(q, Wq, qp, qph);
  cvt_bf16_kernel<<<64, 256, 0, stream>>>(k, kbf, SEQ * DM / 8);
  cvt_bf16_kernel<<<64, 256, 0, stream>>>(v, vbf, SEQ * DM / 8);

  hyper_gemm_kernel<<<4096, 256, 0, stream>>>(qph, Wk, Pk);
  hyper_gemm_kernel<<<4096, 256, 0, stream>>>(qph, Wv, Pv);

  const int smem = 2 * 256 * LDSROW * (int)sizeof(__bf16);  // 270336 B < 320 KB
  (void)hipFuncSetAttribute((const void*)cube_attn_kernel,
                            hipFuncAttributeMaxDynamicSharedMemorySize, smem);
  cube_attn_kernel<<<512, 256, smem, stream>>>(qp, kbf, vbf, Pk, Pv, out);
}
